// LinearCrossAttention_41480794145303
// MI455X (gfx1250) — matrix operand order repaired and checked by reading, not compiled, not run on hardware
//
#include <hip/hip_runtime.h>

// ---------------------------------------------------------------------------
// LinearCrossAttention fused via  final = Wo·BlockDiag(Cᵀ)·(scale·Wq)·x + bo
// All GEMMs in bf16 WMMA (v_wmma_f32_16x16x32_bf16), fp32 accumulate.
// Operands staged via LDS as bf16, except kernel 2 and kernel 3b's B (global); bf16 fragments are 16B
// pairs. __launch_bounds__(.,1) relaxes the VGPR budget so all accumulators
// stay resident (no scratch spills in the K-loop).
// ---------------------------------------------------------------------------

typedef __attribute__((ext_vector_type(16))) __bf16 v16bf;
typedef __attribute__((ext_vector_type(8)))  __bf16 v8bf;
typedef __attribute__((ext_vector_type(8)))  float  v8f;

static constexpr int Bsz = 16;
static constexpr int Nn  = 4096;
static constexpr int Mm  = 1024;
static constexpr int CIN = 512;   // INPUT_DIM == COND_DIM
static constexpr int HID = 512;   // HEADS * DIM_HEAD

__device__ inline v16bf cat8(v8bf lo, v8bf hi) {
  return __builtin_shufflevector(lo, hi, 0,1,2,3,4,5,6,7,8,9,10,11,12,13,14,15);
}

// A fragment (16x32 bf16). src row-major [m][k], base at (m0,k0), k0 % 8 == 0.
// Lane l: row m0+(l&15); 16 contiguous K starting at k0 + (l>=16)*16.
__device__ inline v16bf frag_a_bf16(const __bf16* base, int ldk) {
  int l = threadIdx.x & 31;
  const __bf16* p = base + (size_t)(l & 15) * ldk + ((l >> 4) << 4);
  v8bf lo = *(const v8bf*)(p);
  v8bf hi = *(const v8bf*)(p + 8);
  return cat8(lo, hi);
}

// B fragment (32x16 bf16). src stored k-contiguous per column: [n][k], ldk%8==0.
// Lane l: column n0+(l&15); 16 contiguous K starting at k0 + (l>=16)*16.
__device__ inline v16bf frag_b_bf16(const __bf16* base, int ldk) {
  int l = threadIdx.x & 31;
  const __bf16* p = base + (size_t)(l & 15) * ldk + ((l >> 4) << 4);
  v8bf lo = *(const v8bf*)(p);
  v8bf hi = *(const v8bf*)(p + 8);
  return cat8(lo, hi);
}

// B fragment from fp32 [n][k]-style source (row stride ldk), converted.
__device__ inline v16bf frag_b_f32(const float* base, int ldk) {
  int l = threadIdx.x & 31;
  const float* p = base + (size_t)(l & 15) * ldk + ((l >> 4) << 4);
  v16bf r;
#pragma unroll
  for (int i = 0; i < 16; ++i) r[i] = (__bf16)p[i];
  return r;
}

#define WMMA_BF16(A, Bf, C) \
  __builtin_amdgcn_wmma_f32_16x16x32_bf16(false, (A), false, (Bf), (short)0, (C), false, false)

// ---------------------------------------------------------------------------
// Kernel 1: KV[b][o][m] = sum_c Wkv[o][c] * c_in[b][c][m]   (o<1024, m<1024)
// grid (mtile=8, otile=8, b=16), 256 thr = 8 waves (4 o x 2 m), tile 128x128.
// ---------------------------------------------------------------------------
__global__ void __launch_bounds__(256, 1)
lca_kv_gemm(const float* __restrict__ c_in,
            const float* __restrict__ Wkv,
            float* __restrict__ KV) {
  __shared__ __bf16 at[128][72];                  // A: Wkv chunk [o][cc]
  __shared__ __bf16 bt[128][72];                  // B: cᵀ chunk  [m][cc]
  const int b = blockIdx.z, ot = blockIdx.y, mt = blockIdx.x;
  const int o0 = ot * 128, m0 = mt * 128;
  const float* cb = c_in + (size_t)b * CIN * Mm;
  const int tid = threadIdx.x, wave = tid >> 5, l = tid & 31;
  const int wo = (wave >> 1) * 32, wn = (wave & 1) * 64;
  v8f acc[2][4] = {};

  for (int k0 = 0; k0 < CIN; k0 += 64) {
    __syncthreads();
    {   // stage B: 64 cc rows x 128 m cols, transposed to [m][cc]
      int cc = tid >> 2, nn0 = (tid & 3) * 32;
      const float* src = cb + (size_t)(k0 + cc) * Mm + m0 + nn0;
      if (k0 + 64 < CIN) __builtin_prefetch(src + (size_t)64 * Mm, 0, 1);
#pragma unroll
      for (int i = 0; i < 32; ++i) bt[nn0 + i][cc] = (__bf16)src[i];
    }
    {   // stage A: 128 o rows x 64 cc cols, row-major [o][cc]
      int o = tid >> 1, koff = (tid & 1) * 32;
      const float* src = Wkv + (size_t)(o0 + o) * CIN + k0 + koff;
#pragma unroll
      for (int i = 0; i < 32; ++i) at[o][koff + i] = (__bf16)src[i];
    }
    __syncthreads();
#pragma unroll
    for (int kk = 0; kk < 64; kk += 32) {
      v16bf a[2], bf[4];
#pragma unroll
      for (int i = 0; i < 2; ++i) a[i] = frag_a_bf16(&at[wo + i * 16][kk], 72);
#pragma unroll
      for (int j = 0; j < 4; ++j) bf[j] = frag_b_bf16(&bt[wn + j * 16][kk], 72);
#pragma unroll
      for (int i = 0; i < 2; ++i)
#pragma unroll
        for (int j = 0; j < 4; ++j)
          acc[i][j] = WMMA_BF16(a[i], bf[j], acc[i][j]);
    }
  }
  float* outp = KV + (size_t)b * 1024 * Mm;
  const int col = l & 15, rb = (l >> 4) << 3;
#pragma unroll
  for (int i = 0; i < 2; ++i)
#pragma unroll
    for (int j = 0; j < 4; ++j) {
      int oc = o0 + wo + i * 16 + rb;
      int m  = m0 + wn + j * 16 + col;
#pragma unroll
      for (int g = 0; g < 8; ++g)
        outp[(size_t)(oc + g) * Mm + m] = acc[i][j][g];
    }
}

// ---------------------------------------------------------------------------
// Kernel 2: per (b,h): softmax(k) over M, then C[d][e] = sum_m p[d][m] v[e][m].
// grid 128 blocks (B*H), 128 thr (4 waves, one 16-row d-tile each).
// Probabilities regenerated in-registers inside the WMMA K-loop.
// ---------------------------------------------------------------------------
__global__ void __launch_bounds__(128, 1)
lca_softmax_context(const float* __restrict__ KV,
                    float* __restrict__ CTX) {
  __shared__ float rmax[64], rinv[64], tmp[128];
  const int bh = blockIdx.x, b = bh >> 3, h = bh & 7;
  const float* kbase = KV + ((size_t)b * 1024 + h * 64) * Mm;
  const float* vbase = KV + ((size_t)b * 1024 + 512 + h * 64) * Mm;
  const int tid = threadIdx.x;

  {   // row-wise max & sum(exp), 2 threads per row
    int d = tid & 63, half = tid >> 6;
    const float* row = kbase + (size_t)d * Mm + half * 512;
    float mx = -INFINITY;
    for (int i = 0; i < 512; ++i) mx = fmaxf(mx, row[i]);
    tmp[tid] = mx;
    __syncthreads();
    float m2 = fmaxf(tmp[d], tmp[d + 64]);
    float s = 0.f;
    for (int i = 0; i < 512; ++i) s += __expf(row[i] - m2);
    __syncthreads();
    tmp[tid] = s;
    __syncthreads();
    if (half == 0) { rmax[d] = m2; rinv[d] = 1.f / (tmp[d] + tmp[d + 64]); }
    __syncthreads();
  }

  const int wave = tid >> 5, l = tid & 31;
  const int d0 = wave * 16;
  v8f acc[4] = {};
  for (int m0 = 0; m0 < Mm; m0 += 32) {
    v16bf a;
    {
      int d  = d0 + (l & 15);
      int mb = m0 + ((l >> 4) << 4);
      const float* p = kbase + (size_t)d * Mm + mb;
      float mx = rmax[d], iv = rinv[d];
#pragma unroll
      for (int i = 0; i < 8; ++i) a[i]     = (__bf16)(__expf(p[i]      - mx) * iv);
#pragma unroll
      for (int i = 0; i < 8; ++i) a[8 + i] = (__bf16)(__expf(p[8 + i] - mx) * iv);
    }
#pragma unroll
    for (int j = 0; j < 4; ++j) {
      v16bf bf = frag_b_f32(vbase + (size_t)(j * 16) * Mm + m0, Mm);
      acc[j] = WMMA_BF16(a, bf, acc[j]);
    }
  }
  float* cout = CTX + (size_t)bh * 64 * 64;
  const int col = l & 15, rb = (l >> 4) << 3;
#pragma unroll
  for (int j = 0; j < 4; ++j)
#pragma unroll
    for (int g = 0; g < 8; ++g)
      cout[(size_t)(d0 + rb + g) * 64 + j * 16 + col] = acc[j][g];
}

// ---------------------------------------------------------------------------
// Kernel 3a: T[j=h*64+e][c] = sum_d C[d][e] * (scale*Wq[h*64+d][c]),
// stored transposed as Tt[b][c][j] (bf16) so the next GEMM gets k-contiguous B.
// grid (ctile=4, h=8, b=16), 128 thr (4 waves), tile 64e x 128c, K=64.
// ---------------------------------------------------------------------------
__global__ void __launch_bounds__(128, 1)
lca_make_T(const float* __restrict__ CTX,
           const float* __restrict__ Wq,
           __bf16* __restrict__ Tt) {
  __shared__ __bf16 ctr[64][72];    // A: Cᵀ  [e][d]
  __shared__ __bf16 wqt[128][72];   // B: (scale·Wq)ᵀ [c][d]
  const int b = blockIdx.z, h = blockIdx.y, c0 = blockIdx.x * 128;
  const int tid = threadIdx.x;
  const float* C = CTX + (size_t)(b * 8 + h) * 64 * 64;
  for (int idx = tid; idx < 64 * 64; idx += 128) {
    int d = idx >> 6, e = idx & 63;
    ctr[e][d] = (__bf16)C[idx];
  }
  const float* wqb = Wq + (size_t)(h * 64) * CIN + c0;
  for (int idx = tid; idx < 64 * 128; idx += 128) {
    int d = idx >> 7, c = idx & 127;
    wqt[c][d] = (__bf16)(wqb[(size_t)d * CIN + c] * 0.125f);   // scale = 64^-0.5
  }
  __syncthreads();

  const int wave = tid >> 5, l = tid & 31, e0 = wave * 16;
  v8f acc[8] = {};
#pragma unroll
  for (int kk = 0; kk < 64; kk += 32) {
    v16bf a = frag_a_bf16(&ctr[e0][kk], 72);
#pragma unroll
    for (int j = 0; j < 8; ++j) {
      v16bf bf = frag_b_bf16(&wqt[j * 16][kk], 72);
      acc[j] = WMMA_BF16(a, bf, acc[j]);
    }
  }
  __bf16* tb = Tt + (size_t)b * 512 * 512;
  const int col = l & 15, rb = (l >> 4) << 3;
#pragma unroll
  for (int j = 0; j < 8; ++j) {
    int c = c0 + j * 16 + col;
    int jr = h * 64 + e0 + rb;        // 8 contiguous j values -> 16B store
    v8bf pk;
#pragma unroll
    for (int g = 0; g < 8; ++g) pk[g] = (__bf16)acc[j][g];
    *(v8bf*)(tb + (size_t)c * 512 + jr) = pk;
  }
}

// ---------------------------------------------------------------------------
// Kernel 3b: E[b][o][c] = sum_j Wo[o][j] * Tt[b][c][j]  (bf16 out, row-major c)
// grid (ctile=4, otile=4, b=16), 256 thr = 8 waves (4 o x 2 c), K=512.
// ---------------------------------------------------------------------------
__global__ void __launch_bounds__(256, 1)
lca_make_E(const float* __restrict__ Wo,
           const __bf16* __restrict__ Tt,
           __bf16* __restrict__ E) {
  __shared__ __bf16 at[128][72];
  const int b = blockIdx.z, o0 = blockIdx.y * 128, c0 = blockIdx.x * 128;
  const int tid = threadIdx.x, wave = tid >> 5, l = tid & 31;
  const int wo = (wave >> 1) * 32, wn = (wave & 1) * 64;
  const __bf16* tb = Tt + (size_t)b * 512 * 512;
  v8f acc[2][4] = {};
  for (int k0 = 0; k0 < 512; k0 += 64) {
    __syncthreads();
    {
      int o = tid >> 1, koff = (tid & 1) * 32;
      const float* src = Wo + (size_t)(o0 + o) * HID + k0 + koff;
#pragma unroll
      for (int i = 0; i < 32; ++i) at[o][koff + i] = (__bf16)src[i];
    }
    __syncthreads();
#pragma unroll
    for (int kk = 0; kk < 64; kk += 32) {
      v16bf a[2], bf[4];
#pragma unroll
      for (int i = 0; i < 2; ++i)
        a[i] = frag_a_bf16(&at[wo + i * 16][kk], 72);
#pragma unroll
      for (int j = 0; j < 4; ++j)
        bf[j] = frag_b_bf16(tb + (size_t)(c0 + wn + j * 16) * 512 + k0 + kk, 512);
#pragma unroll
      for (int i = 0; i < 2; ++i)
#pragma unroll
        for (int j = 0; j < 4; ++j)
          acc[i][j] = WMMA_BF16(a[i], bf[j], acc[i][j]);
    }
  }
  __bf16* eb = E + (size_t)b * 512 * 512;
  const int col = l & 15, rb = (l >> 4) << 3;
#pragma unroll
  for (int i = 0; i < 2; ++i)
#pragma unroll
    for (int j = 0; j < 4; ++j) {
      int o = o0 + wo + i * 16 + rb;
      int c = c0 + wn + j * 16 + col;
#pragma unroll
      for (int g = 0; g < 8; ++g)
        eb[(size_t)(o + g) * 512 + c] = (__bf16)acc[i][j][g];
    }
}

// ---------------------------------------------------------------------------
// Kernel 4: out[b][o][n] = sum_c E[b][o][c] * x[b][c][n] + bo[o]
// grid (ntile=32, otile=4, b=16), 256 thr = 8 waves (4 o x 2 n), tile 128x128.
// ---------------------------------------------------------------------------
__global__ void __launch_bounds__(256, 1)
lca_final_gemm(const float* __restrict__ x,
               const __bf16* __restrict__ E,
               const float* __restrict__ bo,
               float* __restrict__ out) {
  __shared__ __bf16 at[128][72];                  // E chunk  [o][c]
  __shared__ __bf16 bt[128][72];                  // xᵀ chunk [n][c]
  const int b = blockIdx.z, o0 = blockIdx.y * 128, n0 = blockIdx.x * 128;
  const float* xb = x + (size_t)b * CIN * Nn;
  const __bf16* eb = E + (size_t)b * 512 * 512;
  const int tid = threadIdx.x, wave = tid >> 5, l = tid & 31;
  const int wo = (wave >> 1) * 32, wn = (wave & 1) * 64;
  v8f acc[2][4] = {};

  for (int k0 = 0; k0 < CIN; k0 += 64) {
    __syncthreads();
    {   // stage B: 64 c rows x 128 n cols -> bt[n][c]
      int cc = tid >> 2, nn0 = (tid & 3) * 32;
      const float* src = xb + (size_t)(k0 + cc) * Nn + n0 + nn0;
      if (k0 + 64 < CIN) __builtin_prefetch(src + (size_t)64 * Nn, 0, 1);
#pragma unroll
      for (int i = 0; i < 32; ++i) bt[nn0 + i][cc] = (__bf16)src[i];
    }
    {   // stage A: 128 o rows x 64 c cols, bf16 16B block copies
      int o = tid >> 1, koff = (tid & 1) * 32;
      const __bf16* src = eb + (size_t)(o0 + o) * 512 + k0 + koff;
      *(v8bf*)&at[o][koff +  0] = *(const v8bf*)(src +  0);
      *(v8bf*)&at[o][koff +  8] = *(const v8bf*)(src +  8);
      *(v8bf*)&at[o][koff + 16] = *(const v8bf*)(src + 16);
      *(v8bf*)&at[o][koff + 24] = *(const v8bf*)(src + 24);
    }
    __syncthreads();
#pragma unroll
    for (int kk = 0; kk < 64; kk += 32) {
      v16bf a[2], bf[4];
#pragma unroll
      for (int i = 0; i < 2; ++i) a[i] = frag_a_bf16(&at[wo + i * 16][kk], 72);
#pragma unroll
      for (int j = 0; j < 4; ++j) bf[j] = frag_b_bf16(&bt[wn + j * 16][kk], 72);
#pragma unroll
      for (int i = 0; i < 2; ++i)
#pragma unroll
        for (int j = 0; j < 4; ++j)
          acc[i][j] = WMMA_BF16(a[i], bf[j], acc[i][j]);
    }
  }
  float* ob = out + (size_t)b * HID * Nn;
  const int col = l & 15, rb = (l >> 4) << 3;
#pragma unroll
  for (int i = 0; i < 2; ++i)
#pragma unroll
    for (int j = 0; j < 4; ++j) {
      int o = o0 + wo + i * 16 + rb;
      int n = n0 + wn + j * 16 + col;
#pragma unroll
      for (int g = 0; g < 8; ++g)
        ob[(size_t)(o + g) * Nn + n] = acc[i][j][g] + bo[o + g];
    }
}

// ---------------------------------------------------------------------------
extern "C" void kernel_launch(void* const* d_in, const int* in_sizes, int n_in,
                              void* d_out, int out_size, void* d_ws, size_t ws_size,
                              hipStream_t stream) {
  const float* x   = (const float*)d_in[0];   // [B,512,4096]
  const float* c   = (const float*)d_in[1];   // [B,512,1024]
  const float* Wq  = (const float*)d_in[2];   // [512,512]
  const float* Wkv = (const float*)d_in[3];   // [1024,512]
  const float* Wo  = (const float*)d_in[4];   // [512,512]
  const float* bo  = (const float*)d_in[5];   // [512]
  float* out = (float*)d_out;                 // [B,512,4096]

  // Workspace layout (bytes):
  //   KV  fp32 [B,1024,1024]  @ 0         (64 MB)
  //   CTX fp32 [B,8,64,64]    @ 64 MB     ( 2 MB)
  //   Tt  bf16 [B,512,512]    @ 66 MB     ( 8 MB)
  //   E   bf16 [B,512,512]    @ 74 MB     ( 8 MB)
  char* ws = (char*)d_ws;
  float*  KV  = (float*)(ws);
  float*  CTX = (float*)(ws + (size_t)67108864);
  __bf16* Tt  = (__bf16*)(ws + (size_t)69206016);
  __bf16* E   = (__bf16*)(ws + (size_t)77594624);

  lca_kv_gemm        <<<dim3(8, 8, Bsz), 256, 0, stream>>>(c, Wkv, KV);
  lca_softmax_context<<<dim3(Bsz * 8),   128, 0, stream>>>(KV, CTX);
  lca_make_T         <<<dim3(4, 8, Bsz), 128, 0, stream>>>(CTX, Wq, Tt);
  lca_make_E         <<<dim3(4, 4, Bsz), 256, 0, stream>>>(Wo, Tt, E);
  lca_final_gemm     <<<dim3(32, 4, Bsz),256, 0, stream>>>(x, E, bo, out);
}
